// ICTDIrrepsE3Conv_26628797235371
// MI455X (gfx1250) — matrix operand order repaired and checked by reading, not compiled, not run on hardware
//
#include <hip/hip_runtime.h>
#include <hip/hip_bf16.h>

typedef __attribute__((ext_vector_type(16))) _Float16 v16h;
typedef __attribute__((ext_vector_type(8)))  _Float16 v8h;
typedef __attribute__((ext_vector_type(8)))  float    v8f;

#define WAVES 4
#define TILE  16

// fast reciprocal: single v_rcp_f32 (~1 ulp), no div_scale/newton chains
__device__ __forceinline__ float fast_rcp(float x) {
  return __builtin_amdgcn_rcpf(x);
}

__device__ __forceinline__ float silu_f(float x) {
  return x * fast_rcp(1.0f + __expf(-x));
}

// A-fragment (16-bit, one 16x32 K-chunk). Lane holds row m = lane&15.
// lanes 0-15: K = [0..7] and [16..23]; lanes 16-31: K = [8..15] and [24..31].
// row points at k=0 of this chunk, k-contiguous storage -> two 16B LDS loads.
__device__ __forceinline__ v16h load_frag_a(const _Float16* row, int lane) {
  int khalf = (lane & 16) ? 8 : 0;
  v8h lo = *(const v8h*)(row + khalf);
  v8h hi = *(const v8h*)(row + 16 + khalf);
  v16h f;
#pragma unroll
  for (int i = 0; i < 8; ++i) { f[i] = lo[i]; f[i + 8] = hi[i]; }
  return f;
}

// B-fragment (16-bit, 32x16 K-chunk), weights stored transposed [n][k].
// lanes 0-15 hold K=[0..7],[16..23], lanes 16-31 hold K=[8..15],[24..31] of column n = lane&15.
__device__ __forceinline__ v16h load_frag_b(const _Float16* coln, int lane) {
  int koff = (lane & 16) ? 8 : 0;
  v8h lo = *(const v8h*)(coln + koff);
  v8h hi = *(const v8h*)(coln + 16 + koff);
  v16h f;
#pragma unroll
  for (int i = 0; i < 8; ++i) { f[i] = lo[i]; f[i + 8] = hi[i]; }
  return f;
}

__device__ __forceinline__ v8f wmma_f16(v16h a, v16h b, v8f c) {
  return __builtin_amdgcn_wmma_f32_16x16x32_f16(false, a, false, b,
                                                (short)0, c, false, false);
}

// ---------------------------------------------------------------- zero init
__global__ void zero_kernel(float* __restrict__ out, long long nOut,
                            float* __restrict__ counts, int N) {
  long long i = (long long)blockIdx.x * blockDim.x + threadIdx.x;
  if (i < nOut) out[i] = 0.0f;
  if (i < N)    counts[i] = 0.0f;
}

// ------------------------------------------------------------- node MLP: Ai
__global__ void node_kernel(const int* __restrict__ A,
                            const float* __restrict__ emb_table,
                            const float* __restrict__ w1, const float* __restrict__ b1,
                            const float* __restrict__ w2, const float* __restrict__ b2,
                            float* __restrict__ Ai, int N) {
  int i = blockIdx.x * blockDim.x + threadIdx.x;
  if (i >= N) return;
  const float* e = emb_table + A[i] * 16;
  float h[64];
#pragma unroll 8
  for (int j = 0; j < 64; ++j) {
    float s = b1[j];
#pragma unroll
    for (int k = 0; k < 16; ++k) s = fmaf(e[k], w1[k * 64 + j], s);
    h[j] = silu_f(s);
  }
#pragma unroll
  for (int u = 0; u < 8; ++u) {
    float s = b2[u];
#pragma unroll 8
    for (int j = 0; j < 64; ++j) s = fmaf(h[j], w2[j * 8 + u], s);
    Ai[(long long)i * 8 + u] = s;
  }
}

// --------------------------------------------------------------- edge tiles
__global__ __launch_bounds__(WAVES * 32) void edge_kernel(
    const float* __restrict__ pos, const int* __restrict__ batch,
    const int* __restrict__ esrc, const int* __restrict__ edst,
    const float* __restrict__ shifts, const float* __restrict__ cell,
    const float* __restrict__ fw1, const float* __restrict__ fb1,
    const float* __restrict__ fw2, const float* __restrict__ fb2,
    const float* __restrict__ fw3, const float* __restrict__ fb3,
    const float* __restrict__ tpw,
    const float* __restrict__ Ai,
    float* __restrict__ out, float* __restrict__ counts, int E) {

  // weights, transposed [n][k], k-contiguous, f16
  __shared__ alignas(16) _Float16 sW1[4 * 16 * 32];   // 16->64, K padded to 32
  __shared__ alignas(16) _Float16 sW2[4 * 16 * 64];   // 64->64
  __shared__ alignas(16) _Float16 sW3[16 * 64];       // 64->16 (15 real cols)
  __shared__ alignas(16) _Float16 sWL[3 * 16 * 64];   // bilinear W_l, [l][w][u*8+v]
  __shared__ float sB1[64], sB2[64], sB3[16];
  // per-wave activation tiles
  __shared__ alignas(16) _Float16 sEmb[WAVES][16 * 32];
  __shared__ alignas(16) _Float16 sH1[WAVES][16 * 64];
  __shared__ alignas(16) _Float16 sH2[WAVES][16 * 64];
  __shared__ alignas(16) _Float16 sOuter[WAVES][16 * 64];
  __shared__ float sY[WAVES][16 * 9];
  __shared__ float sGate[WAVES][16 * 3];
  __shared__ int   sDst[WAVES][16];

  const int tid  = threadIdx.x;
  const int lane = tid & 31;
  const int wv   = tid >> 5;
  const int nthr = blockDim.x;
  const int tileBase = (blockIdx.x * WAVES + wv) * TILE;

  // ---- stage weights into LDS (f32 -> f16, transposed)
  const int PIDX0 = 0, PIDX1 = 3, PIDX2 = 9;
  for (int i = tid; i < 4 * 16 * 32; i += nthr) {
    int t = i >> 9, rem = i & 511, n = rem >> 5, k = rem & 31;
    sW1[i] = (_Float16)((k < 16) ? fw1[k * 64 + 16 * t + n] : 0.0f);
  }
  for (int i = tid; i < 4 * 16 * 64; i += nthr) {
    int t = i >> 10, rem = i & 1023, n = rem >> 6, k = rem & 63;
    sW2[i] = (_Float16)fw2[k * 64 + 16 * t + n];
  }
  for (int i = tid; i < 16 * 64; i += nthr) {
    int n = i >> 6, k = i & 63;
    sW3[i] = (_Float16)((n < 15) ? fw3[k * 15 + n] : 0.0f);
  }
  for (int i = tid; i < 3 * 16 * 64; i += nthr) {
    int L = i >> 10, rem = i & 1023, w = rem >> 6, k = rem & 63;
    int p = (L == 0) ? PIDX0 : (L == 1) ? PIDX1 : PIDX2;
    // tp_w[p][u][v][w], u = k>>3, v = k&7
    sWL[i] = (_Float16)tpw[(((p * 8 + (k >> 3)) * 8) + (k & 7)) * 16 + w];
  }
  for (int i = tid; i < 64; i += nthr) { sB1[i] = fb1[i]; sB2[i] = fb2[i]; }
  for (int i = tid; i < 16; i += nthr) { sB3[i] = (i < 15) ? fb3[i] : 0.0f; }
  __syncthreads();

  // ---- phase 1: per-edge geometry / radial basis / outer product (lanes 0..15)
  if (lane < 16) {
    const int m = lane;
    const int e = tileBase + m;
    const bool valid = (e < E);
    const int ec  = valid ? e : 0;
    const int src = esrc[ec];
    const int dst = edst[ec];
    const float sx = shifts[(long long)ec * 3 + 0];
    const float sy = shifts[(long long)ec * 3 + 1];
    const float sz = shifts[(long long)ec * 3 + 2];
    const float* C = cell + (long long)batch[src] * 9;
    float sv0 = sx * C[0] + sy * C[3] + sz * C[6];
    float sv1 = sx * C[1] + sy * C[4] + sz * C[7];
    float sv2 = sx * C[2] + sy * C[5] + sz * C[8];
    float vx = pos[(long long)dst * 3 + 0] - pos[(long long)src * 3 + 0] + sv0;
    float vy = pos[(long long)dst * 3 + 1] - pos[(long long)src * 3 + 1] + sv1;
    float vz = pos[(long long)dst * 3 + 2] - pos[(long long)src * 3 + 2] + sv2;
    float len = __builtin_sqrtf(vx * vx + vy * vy + vz * vz);
    float inv = fast_rcp(fmaxf(len, 1e-8f));
    float nx = vx * inv, ny = vy * inv, nz = vz * inv;

    const float s3 = 1.7320508075688772f;   // sqrt(3)
    const float s15 = 3.872983346207417f;   // sqrt(15)
    const float s5h = 1.118033988749895f;   // sqrt(5)/2
    float* Yw = &sY[wv][m * 9];
    Yw[0] = 1.0f;
    Yw[1] = s3 * nx; Yw[2] = s3 * ny; Yw[3] = s3 * nz;
    Yw[4] = s15 * nx * ny;
    Yw[5] = s15 * ny * nz;
    Yw[6] = s5h * (3.0f * nz * nz - 1.0f);
    Yw[7] = s15 * nx * nz;
    Yw[8] = 0.5f * s15 * (nx * nx - ny * ny);

    // radial basis: centers 5*(k+1)/17, step 5/17, * sqrt(16)/1.12
    _Float16* embRow = &sEmb[wv][m * 32];
#pragma unroll
    for (int k = 0; k < 16; ++k) {
      float v = 5.0f * (float)(k + 1) * (1.0f / 17.0f);
      float d = (len - v) * (17.0f / 5.0f);
      embRow[k] = (_Float16)(__expf(-d * d) * (4.0f / 1.12f));
    }
#pragma unroll
    for (int k = 16; k < 32; ++k) embRow[k] = (_Float16)0.0f;

    // outer(Ai_src, x2_0) -> 64-vector
    const float* as = Ai + (long long)src * 8;
    const float* ad = Ai + (long long)dst * 8;
    _Float16* orow = &sOuter[wv][m * 64];
#pragma unroll
    for (int u = 0; u < 8; ++u) {
      float au = as[u];
#pragma unroll
      for (int v = 0; v < 8; ++v)
        orow[u * 8 + v] = (_Float16)(au * ad[v]);
    }
    sDst[wv][m] = dst;
    if (valid) atomicAdd(&counts[dst], 1.0f);
  }
  __syncthreads();

  const int n   = lane & 15;
  const int hi8 = (lane & 16) ? 8 : 0;

  // ---- layer 1: emb(16x32) @ W1(32x64), silu -> sH1
  {
    v16h a = load_frag_a(&sEmb[wv][(lane & 15) * 32], lane);
#pragma unroll
    for (int t = 0; t < 4; ++t) {
      v16h b = load_frag_b(&sW1[(t * 16 + n) * 32], lane);
      v8f c = {};
      c = wmma_f16(a, b, c);
      const int col = t * 16 + n;
      const float bias = sB1[col];
#pragma unroll
      for (int r = 0; r < 8; ++r)
        sH1[wv][(r + hi8) * 64 + col] = (_Float16)silu_f(c[r] + bias);
    }
  }
  __syncthreads();

  // ---- layer 2: h1(16x64) @ W2(64x64), silu -> sH2
  {
    const _Float16* aRow = &sH1[wv][(lane & 15) * 64];
    v16h a0 = load_frag_a(aRow, lane);
    v16h a1 = load_frag_a(aRow + 32, lane);
#pragma unroll
    for (int t = 0; t < 4; ++t) {
      const _Float16* bc = &sW2[(t * 16 + n) * 64];
      v8f c = {};
      c = wmma_f16(a0, load_frag_b(bc, lane), c);
      c = wmma_f16(a1, load_frag_b(bc + 32, lane), c);
      const int col = t * 16 + n;
      const float bias = sB2[col];
#pragma unroll
      for (int r = 0; r < 8; ++r)
        sH2[wv][(r + hi8) * 64 + col] = (_Float16)silu_f(c[r] + bias);
    }
  }
  __syncthreads();

  // ---- layer 3: h2(16x64) @ W3(64x16) -> gates (cols 0,3,9)
  {
    const _Float16* aRow = &sH2[wv][(lane & 15) * 64];
    v16h a0 = load_frag_a(aRow, lane);
    v16h a1 = load_frag_a(aRow + 32, lane);
    const _Float16* bc = &sW3[n * 64];
    v8f c = {};
    c = wmma_f16(a0, load_frag_b(bc, lane), c);
    c = wmma_f16(a1, load_frag_b(bc + 32, lane), c);
    const int which = (n == 0) ? 0 : (n == 3) ? 1 : (n == 9) ? 2 : -1;
    if (which >= 0) {
      const float bias = sB3[n];
#pragma unroll
      for (int r = 0; r < 8; ++r)
        sGate[wv][(r + hi8) * 3 + which] = c[r] + bias;
    }
  }
  __syncthreads();

  // ---- bilinear tensor product + gated scatter into out
  {
    const _Float16* aRow = &sOuter[wv][(lane & 15) * 64];
    v16h a0 = load_frag_a(aRow, lane);
    v16h a1 = load_frag_a(aRow + 32, lane);
    const int offs[3]  = {0, 16, 64};
    const int ybase[3] = {0, 1, 4};
#pragma unroll
    for (int L = 0; L < 3; ++L) {
      const _Float16* bc = &sWL[(L * 16 + n) * 64];
      v8f c = {};
      c = wmma_f16(a0, load_frag_b(bc, lane), c);
      c = wmma_f16(a1, load_frag_b(bc + 32, lane), c);
      const int nw = 2 * L + 1;
#pragma unroll
      for (int r = 0; r < 8; ++r) {
        const int m = r + hi8;
        if (tileBase + m < E) {
          const float val = c[r] * 0.125f * sGate[wv][m * 3 + L];  // PATH_NORM = 1/8
          float* o = out + (long long)sDst[wv][m] * 144 + offs[L] + n * nw;
          const float* Yw = &sY[wv][m * 9 + ybase[L]];
          for (int i = 0; i < nw; ++i)
            atomicAdd(o + i, val * Yw[i]);
        }
      }
    }
  }
}

// ------------------------------------------------------------ segment mean
__global__ void div_kernel(float* __restrict__ out,
                           const float* __restrict__ counts, int N) {
  int i = blockIdx.x * blockDim.x + threadIdx.x;
  if (i < N * 144)
    out[i] *= fast_rcp(fmaxf(counts[i / 144], 1.0f));
}

extern "C" void kernel_launch(void* const* d_in, const int* in_sizes, int n_in,
                              void* d_out, int out_size, void* d_ws, size_t ws_size,
                              hipStream_t stream) {
  const float* pos       = (const float*)d_in[0];
  const int*   A         = (const int*)d_in[1];
  const int*   batch     = (const int*)d_in[2];
  const int*   esrc      = (const int*)d_in[3];
  const int*   edst      = (const int*)d_in[4];
  const float* shifts    = (const float*)d_in[5];
  const float* cell      = (const float*)d_in[6];
  const float* emb_table = (const float*)d_in[7];
  const float* aw1 = (const float*)d_in[8];
  const float* ab1 = (const float*)d_in[9];
  const float* aw2 = (const float*)d_in[10];
  const float* ab2 = (const float*)d_in[11];
  const float* fw1 = (const float*)d_in[12];
  const float* fb1 = (const float*)d_in[13];
  const float* fw2 = (const float*)d_in[14];
  const float* fb2 = (const float*)d_in[15];
  const float* fw3 = (const float*)d_in[16];
  const float* fb3 = (const float*)d_in[17];
  const float* tpw = (const float*)d_in[18];

  const int N = in_sizes[1];
  const int E = in_sizes[3];
  float* out    = (float*)d_out;
  float* Ai     = (float*)d_ws;                 // N*8 floats
  float* counts = Ai + (long long)N * 8;        // N floats

  const long long nOut = (long long)N * 144;
  zero_kernel<<<(unsigned)((nOut + 255) / 256), 256, 0, stream>>>(out, nOut, counts, N);
  node_kernel<<<(N + 255) / 256, 256, 0, stream>>>(A, emb_table, aw1, ab1, aw2, ab2, Ai, N);
  const int edgesPerBlock = WAVES * TILE;  // 64
  edge_kernel<<<(E + edgesPerBlock - 1) / edgesPerBlock, WAVES * 32, 0, stream>>>(
      pos, batch, esrc, edst, shifts, cell,
      fw1, fb1, fw2, fb2, fw3, fb3, tpw, Ai, out, counts, E);
  div_kernel<<<(int)((nOut + 255) / 256), 256, 0, stream>>>(out, counts, N);
}
